// Answer_Decoder_4140348473528
// MI455X (gfx1250) — matrix operand order repaired and checked by reading, not compiled, not run on hardware
//
#include <hip/hip_runtime.h>

// ---------------------------------------------------------------------------
// LSTM answer decoder for MI455X (gfx1250), bf16 WMMA pipeline.
//
// Math notes (faithful to reference):
//  * softmax over a length-1 axis == 1.0 exactly, so ctx == concat(q, img)
//    every step; attn_W/attn_b/v_W contribute nothing and are skipped.
//  * Layer-0 input projection (cat(embed_t, ctx) @ W_ih0.T + b_ih0 + b_hh0)
//    is time-independent given the token sequence -> one big GEMM up front.
//  * FC head deferred: collect h2 for all (t,b) then one [1536,512]x[512,32000]
//    GEMM (output write 196.6 MB dominates; fc_W read once).
//  * FC A-block staged in LDS via gfx1250 async LDS copy to kill the 8x
//    redundant per-wave A reads from L2 (3 GB -> 0.38 GB).
// ---------------------------------------------------------------------------

typedef __attribute__((ext_vector_type(16))) __bf16 v16bf;
typedef __attribute__((ext_vector_type(8)))  float  v8f;
typedef unsigned short u16t;
typedef unsigned int   u32t;

union ABu { v16bf v; uint4 q[2]; };

__device__ inline u16t f2bf(float f) {
  union { float f; u32t u; } x; x.f = f;
  u32t u = x.u;
  u32t r = u + 0x7FFFu + ((u >> 16) & 1u);          // round-to-nearest-even
  if ((u & 0x7F800000u) == 0x7F800000u) r = u;      // Inf/NaN passthrough
  return (u16t)(r >> 16);
}

__device__ inline float sigm(float x) { return 1.0f / (1.0f + __expf(-x)); }

__device__ inline v8f wmma_bf16(ABu a, ABu b, v8f c) {
  // D = A(16x32 bf16) * B(32x16 bf16) + C(16x16 f32)
  return __builtin_amdgcn_wmma_f32_16x16x32_bf16(
      false, a.v, false, b.v, (short)0, c, false, false);
}

// ---------------------------------------------------------------------------
// Generic 64x16 output strip (4 M-tiles x 1 N-tile) GEMM accumulator.
//  A: row-major [.., lda] bf16, pre-offset to the 64-row block.
//  Brow: row n of W (row-major [N, lda] bf16) -> B column n of A*W^T.
// ---------------------------------------------------------------------------
__device__ inline void gemm_strip4(v8f acc[4], const u16t* __restrict__ A, int lda,
                                   const u16t* __restrict__ Brow, int kIter, int lane) {
  const int half = lane >> 4;
  const int mr   = lane & 15;
  for (int ks = 0; ks < kIter; ++ks) {
    const int k0 = ks * 32;
    ABu b;
    const u16t* bp = Brow + k0 + half * 8;    // K = k0+half*8..+7 and k0+16+half*8..+7
    b.q[0] = *(const uint4*)(bp);
    b.q[1] = *(const uint4*)(bp + 16);
#pragma unroll
    for (int i = 0; i < 4; ++i) {
      const u16t* ap = A + (size_t)(i * 16 + mr) * lda + k0 + half * 8;
      ABu a;
      a.q[0] = *(const uint4*)(ap);           // K = k0+half*8 .. +7
      a.q[1] = *(const uint4*)(ap + 16);      // K = k0+16+half*8 .. +7
      acc[i] = wmma_bf16(a, b, acc[i]);
    }
  }
}

// ---------------------------------------------------------------------------
// Layer-0 input projection: G0[m, n] = X[m,:]@W_ih0[n,:] + b_ih0[n] + b_hh0[n]
//   X: [1536, 1280] bf16 (row m = t*64+b), W: [2048, 1280] bf16.
// ---------------------------------------------------------------------------
__global__ __launch_bounds__(256)
void g0_gemm_kernel(const u16t* __restrict__ X, const u16t* __restrict__ W,
                    const float* __restrict__ bih, const float* __restrict__ bhh,
                    float* __restrict__ G0) {
  const int lane = threadIdx.x & 31;
  const int wv   = threadIdx.x >> 5;
  const int n0   = (blockIdx.x * 8 + wv) * 16;
  const int m0   = blockIdx.y * 64;
  const int half = lane >> 4, nc = lane & 15;
  v8f acc[4];
#pragma unroll
  for (int i = 0; i < 4; ++i)
#pragma unroll
    for (int r = 0; r < 8; ++r) acc[i][r] = 0.f;
  gemm_strip4(acc, X + (size_t)m0 * 1280, 1280,
              W + (size_t)(n0 + nc) * 1280, 40, lane);
  const float bias = bih[n0 + nc] + bhh[n0 + nc];
#pragma unroll
  for (int i = 0; i < 4; ++i)
#pragma unroll
    for (int r = 0; r < 8; ++r) {
      const int m = m0 + i * 16 + half * 8 + r;
      G0[(size_t)m * 2048 + n0 + nc] = acc[i][r] + bias;
    }
}

// ---------------------------------------------------------------------------
// One LSTM cell for one timestep. Grid: 16 blocks x 256 thr = 128 waves.
// Wave computes one 16-row M-tile x the same 16 cols of all 4 gates, so the
// LSTM pointwise update runs in-register. Layer 0: input GEMM precomputed
// in G0 (used as C init); layers 1/2: two K=512 GEMMs (cur@Wih^T, hprev@Whh^T).
// ---------------------------------------------------------------------------
__global__ __launch_bounds__(256)
void lstm_cell_kernel(const float* __restrict__ G0,    // layer0 only, else null
                      const u16t* __restrict__ Wih,    // [2048,512] bf16 (null for l0)
                      const u16t* __restrict__ Whh,    // [2048,512] bf16
                      const float* __restrict__ bih, const float* __restrict__ bhh,
                      const u16t* __restrict__ cur,    // [64,512] bf16 (null for l0)
                      const u16t* __restrict__ hprev,  // [64,512] bf16
                      u16t* __restrict__ hout,         // [64,512] bf16
                      float* __restrict__ cst,         // [64,512] f32 in/out
                      u16t* __restrict__ h2out,        // [1536,512] bf16 (layer2)
                      int t) {
  const int lane  = threadIdx.x & 31;
  const int w     = (blockIdx.x << 3) + (threadIdx.x >> 5);
  const int m0    = (w & 3) << 4;     // 4 M-tiles cover B=64
  const int n0    = (w >> 2) << 4;    // 32 col-groups cover H=512
  const int half  = lane >> 4;
  const int mr    = lane & 15;
  const int nc    = lane & 15;

  v8f acc[4];                         // gates i, f, g, o
  if (G0) {
    const float* g0p = G0 + (size_t)(t * 64 + m0 + half * 8) * 2048 + n0 + nc;
#pragma unroll
    for (int g = 0; g < 4; ++g)
#pragma unroll
      for (int r = 0; r < 8; ++r)
        acc[g][r] = g0p[(size_t)r * 2048 + g * 512];
  } else {
#pragma unroll
    for (int g = 0; g < 4; ++g) {
      const float bv = bih[g * 512 + n0 + nc] + bhh[g * 512 + n0 + nc];
#pragma unroll
      for (int r = 0; r < 8; ++r) acc[g][r] = bv;
    }
  }

  const u16t* Asrc[2] = { cur,  hprev };
  const u16t* Wsrc[2] = { Wih,  Whh   };
  for (int p = 0; p < 2; ++p) {
    const u16t* A = Asrc[p];
    const u16t* W = Wsrc[p];
    if (!A || !W) continue;           // uniform branch (layer 0 skips ih)
    const u16t* arow = A + (size_t)(m0 + mr) * 512;
    for (int ks = 0; ks < 16; ++ks) {
      const int k0 = ks * 32;
      ABu a;
      a.q[0] = *(const uint4*)(arow + k0 + half * 8);
      a.q[1] = *(const uint4*)(arow + k0 + 16 + half * 8);
#pragma unroll
      for (int g = 0; g < 4; ++g) {
        const u16t* brow = W + (size_t)(g * 512 + n0 + nc) * 512 + k0 + half * 8;
        ABu b;
        b.q[0] = *(const uint4*)(brow);
        b.q[1] = *(const uint4*)(brow + 16);
        acc[g] = wmma_bf16(a, b, acc[g]);
      }
    }
  }

  // Pointwise LSTM update (torch gate order i, f, g, o)
#pragma unroll
  for (int r = 0; r < 8; ++r) {
    const int m = m0 + half * 8 + r;
    const size_t off = (size_t)m * 512 + n0 + nc;
    const float iv = sigm(acc[0][r]);
    const float fv = sigm(acc[1][r]);
    const float gv = tanhf(acc[2][r]);
    const float ov = sigm(acc[3][r]);
    const float cn = fv * cst[off] + iv * gv;
    const float hn = ov * tanhf(cn);
    cst[off]  = cn;
    hout[off] = f2bf(hn);
    if (h2out) h2out[(size_t)(t * 64 + m) * 512 + n0 + nc] = f2bf(hn);
  }
}

// ---------------------------------------------------------------------------
// FC head: out[b, t, :] = H2[t*64+b, :] @ fc_W^T + fc_b
//   H2: [1536, 512] bf16, fc_W: [32000, 512] bf16, out f32 [64, 24, 32000].
// A-block [64,512] (64 KB) staged ONCE per workgroup into LDS via gfx1250
// async LDS copies, then all 8 waves feed WMMA A-operands from LDS.
// ---------------------------------------------------------------------------
__global__ __launch_bounds__(256)
void fc_gemm_kernel(const u16t* __restrict__ H2, const u16t* __restrict__ W,
                    const float* __restrict__ bias, float* __restrict__ out) {
  __shared__ alignas(16) u16t sA[64 * 512];        // 64 KB of the 320 KB LDS
  const int tid  = threadIdx.x;
  const int lane = tid & 31;
  const int wv   = tid >> 5;
  const int n0   = (blockIdx.x * 8 + wv) * 16;
  const int m0   = blockIdx.y * 64;
  const int half = lane >> 4, mr = lane & 15, nc = lane & 15;

  // ---- async stage A block: 256 threads x 16 chunks x 16B = 64 KB --------
  {
    const u16t* src = H2 + (size_t)m0 * 512;
#pragma unroll
    for (int j = 0; j < 16; ++j) {
      const int elem = (tid + j * 256) * 8;        // 8 bf16 = 16 bytes/chunk
      // ISA aperture rule: LDS offset = flat_addr[31:0]
      const unsigned int ldsoff = (unsigned int)(size_t)(&sA[elem]);
      const u16t* g = src + elem;
      asm volatile("global_load_async_to_lds_b128 %0, %1, off"
                   :: "v"(ldsoff), "v"(g) : "memory");
    }
    asm volatile("s_wait_asynccnt 0x0" ::: "memory");
  }
  __syncthreads();

  // ---- GEMM: B streamed from L2, A from LDS ------------------------------
  v8f acc[4];
#pragma unroll
  for (int i = 0; i < 4; ++i)
#pragma unroll
    for (int r = 0; r < 8; ++r) acc[i][r] = 0.f;

  const u16t* Brow = W + (size_t)(n0 + nc) * 512;
  for (int ks = 0; ks < 16; ++ks) {
    const int k0 = ks * 32;
    ABu b;
    const u16t* bp = Brow + k0 + half * 8;
    b.q[0] = *(const uint4*)(bp);
    b.q[1] = *(const uint4*)(bp + 16);
#pragma unroll
    for (int i = 0; i < 4; ++i) {
      const u16t* ap = &sA[(i * 16 + mr) * 512 + k0 + half * 8];
      ABu a;
      a.q[0] = *(const uint4*)(ap);               // ds_load_b128
      a.q[1] = *(const uint4*)(ap + 16);
      acc[i] = wmma_bf16(a, b, acc[i]);
    }
  }

  const float bv = bias[n0 + nc];
#pragma unroll
  for (int i = 0; i < 4; ++i)
#pragma unroll
    for (int r = 0; r < 8; ++r) {
      const int m = m0 + i * 16 + half * 8 + r;  // m = t*64 + b
      const int b = m & 63, tt = m >> 6;
      out[((size_t)b * 24 + tt) * 32000 + n0 + nc] = acc[i][r] + bv;
    }
}

// --------------------------- small prep kernels ----------------------------
__global__ void cvt_f32_bf16_kernel(const float* __restrict__ s,
                                    u16t* __restrict__ d, int n) {
  for (int i = blockIdx.x * blockDim.x + threadIdx.x; i < n;
       i += gridDim.x * blockDim.x)
    d[i] = f2bf(s[i]);
}

// X[t*64+b, :] = cat(embedding[seq[b,t]] (256), q[b] (512), img[b] (512)) bf16
__global__ void build_x_kernel(const float* __restrict__ qf,
                               const float* __restrict__ imf,
                               const int* __restrict__ seq,
                               const float* __restrict__ emb,
                               u16t* __restrict__ X) {
  const int row = blockIdx.x;
  const int t = row >> 6, b = row & 63;
  const int tok = seq[b * 24 + t];
  const float* e = emb + (size_t)tok * 256;
  for (int c = threadIdx.x; c < 1280; c += blockDim.x) {
    const float v = (c < 256) ? e[c]
                  : (c < 768) ? qf[b * 512 + c - 256]
                              : imf[b * 512 + c - 768];
    X[(size_t)row * 1280 + c] = f2bf(v);
  }
}

__global__ void zero_state_kernel(u16t* __restrict__ hbf, float* __restrict__ cst) {
  const int n_h = 2 * 3 * 64 * 512;
  const int n_c = 3 * 64 * 512;
  for (int i = blockIdx.x * blockDim.x + threadIdx.x; i < n_h;
       i += gridDim.x * blockDim.x) hbf[i] = 0;
  for (int i = blockIdx.x * blockDim.x + threadIdx.x; i < n_c;
       i += gridDim.x * blockDim.x) cst[i] = 0.f;
}

// ---------------------------------------------------------------------------
extern "C" void kernel_launch(void* const* d_in, const int* in_sizes, int n_in,
                              void* d_out, int out_size, void* d_ws, size_t ws_size,
                              hipStream_t stream) {
  (void)in_sizes; (void)n_in; (void)out_size; (void)ws_size;

  const float* qf   = (const float*)d_in[0];
  const float* imf  = (const float*)d_in[1];
  const int*   seq  = (const int*)  d_in[2];
  const float* emb  = (const float*)d_in[3];
  // d_in[4..6] = attn_W, attn_b, v_W: softmax over a singleton axis == 1.0,
  // so the attention contributes exactly ctx = concat(q, img). Skipped.
  const float* fcW  = (const float*)d_in[7];
  const float* fcb  = (const float*)d_in[8];
  const float* Wih_f[3] = {(const float*)d_in[9],  (const float*)d_in[13], (const float*)d_in[17]};
  const float* Whh_f[3] = {(const float*)d_in[10], (const float*)d_in[14], (const float*)d_in[18]};
  const float* bih[3]   = {(const float*)d_in[11], (const float*)d_in[15], (const float*)d_in[19]};
  const float* bhh[3]   = {(const float*)d_in[12], (const float*)d_in[16], (const float*)d_in[20]};
  float* out = (float*)d_out;

  char* ws = (char*)d_ws;
  size_t off = 0;
  auto take = [&](size_t bytes) -> char* {
    char* p = ws + off;
    off += (bytes + 255) & ~(size_t)255;
    return p;
  };

  u16t*  Xbf   = (u16t*) take((size_t)1536 * 1280 * 2);
  u16t*  Wih0b = (u16t*) take((size_t)2048 * 1280 * 2);
  u16t*  Wih1b = (u16t*) take((size_t)2048 * 512 * 2);
  u16t*  Wih2b = (u16t*) take((size_t)2048 * 512 * 2);
  u16t*  Whh0b = (u16t*) take((size_t)2048 * 512 * 2);
  u16t*  Whh1b = (u16t*) take((size_t)2048 * 512 * 2);
  u16t*  Whh2b = (u16t*) take((size_t)2048 * 512 * 2);
  u16t*  fcWb  = (u16t*) take((size_t)32000 * 512 * 2);
  float* G0    = (float*)take((size_t)1536 * 2048 * 4);
  u16t*  hbf   = (u16t*) take((size_t)2 * 3 * 64 * 512 * 2);   // h double-buffer
  float* cst   = (float*)take((size_t)3 * 64 * 512 * 4);       // cell state
  u16t*  H2bf  = (u16t*) take((size_t)1536 * 512 * 2);         // collected h2

  // Weight conversions f32 -> bf16 (once per call; all fit in L2 after this)
  cvt_f32_bf16_kernel<<<2048, 256, 0, stream>>>(Wih_f[0], Wih0b, 2048 * 1280);
  cvt_f32_bf16_kernel<<<1024, 256, 0, stream>>>(Whh_f[0], Whh0b, 2048 * 512);
  cvt_f32_bf16_kernel<<<1024, 256, 0, stream>>>(Wih_f[1], Wih1b, 2048 * 512);
  cvt_f32_bf16_kernel<<<1024, 256, 0, stream>>>(Whh_f[1], Whh1b, 2048 * 512);
  cvt_f32_bf16_kernel<<<1024, 256, 0, stream>>>(Wih_f[2], Wih2b, 2048 * 512);
  cvt_f32_bf16_kernel<<<1024, 256, 0, stream>>>(Whh_f[2], Whh2b, 2048 * 512);
  cvt_f32_bf16_kernel<<<4096, 256, 0, stream>>>(fcW, fcWb, 32000 * 512);

  build_x_kernel<<<1536, 256, 0, stream>>>(qf, imf, seq, emb, Xbf);
  zero_state_kernel<<<384, 256, 0, stream>>>(hbf, cst);

  // Precompute layer-0 input projection for all 24 steps: [1536,2048]
  g0_gemm_kernel<<<dim3(16, 24), 256, 0, stream>>>(Xbf, Wih0b, bih[0], bhh[0], G0);

  const u16t* WihB[3] = { nullptr, Wih1b, Wih2b };
  const u16t* WhhB[3] = { Whh0b,   Whh1b, Whh2b };

  // Sequential recurrence: 24 steps x 3 layers, h double-buffered by parity.
  for (int t = 0; t < 24; ++t) {
    for (int l = 0; l < 3; ++l) {
      u16t* hprev = hbf + (((t) & 1) * 3 + l) * (64 * 512);
      u16t* hcur  = hbf + (((t + 1) & 1) * 3 + l) * (64 * 512);
      u16t* curin = (l == 0) ? nullptr
                             : hbf + (((t + 1) & 1) * 3 + (l - 1)) * (64 * 512);
      float* cl   = cst + l * (64 * 512);
      lstm_cell_kernel<<<16, 256, 0, stream>>>(
          (l == 0) ? G0 : nullptr,
          WihB[l], WhhB[l],
          (l == 0) ? nullptr : bih[l],
          (l == 0) ? nullptr : bhh[l],
          curin, hprev, hcur, cl,
          (l == 2) ? H2bf : nullptr, t);
    }
  }

  // Deferred FC head: one GEMM, output-write bound (~197 MB @ 23.3 TB/s)
  fc_gemm_kernel<<<dim3(250, 24), 256, 0, stream>>>(H2bf, fcWb, fcb, out);
}
